// triGAT_without_edge_feature_41781441856241
// MI455X (gfx1250) — matrix operand order repaired and checked by reading, not compiled, not run on hardware
//
#include <hip/hip_runtime.h>
#include <hip/hip_bf16.h>

typedef __bf16 v16bf __attribute__((ext_vector_type(16)));
typedef float  v8f   __attribute__((ext_vector_type(8)));

#define NEG_SLOPE 0.2f

static __device__ __forceinline__ __bf16 f2bf(float f) {
    union { float f; unsigned u; } v; v.f = f;
    unsigned r = v.u + 0x7fffu + ((v.u >> 16) & 1u);   // round-to-nearest-even
    union { unsigned short s; __bf16 b; } o; o.s = (unsigned short)(r >> 16);
    return o.b;
}

static __device__ __forceinline__ float leaky(float x) {
    return x > 0.f ? x : NEG_SLOPE * x;
}

static __device__ __forceinline__ void atomicMaxFloat(float* addr, float val) {
    if (val >= 0.f) atomicMax((int*)addr, __float_as_int(val));
    else            atomicMin((unsigned int*)addr, __float_as_uint(val));
}

// ---------------- conversion kernels ----------------
__global__ void k_cvt_x(const float* __restrict__ X, __bf16* __restrict__ Xb, int n) {
    int t = blockIdx.x * blockDim.x + threadIdx.x;
    if (t < n) Xb[t] = f2bf(X[t]);
}

// Wt[br][col][k] = W_br[k*128 + col]  (col-major so each lane reads contiguous K)
__global__ void k_cvt_w(const float* __restrict__ W1, const float* __restrict__ W2,
                        const float* __restrict__ W3, __bf16* __restrict__ Wt) {
    int t = blockIdx.x * blockDim.x + threadIdx.x;
    if (t >= 3 * 128 * 64) return;
    int br = t / (128 * 64);
    int r  = t % (128 * 64);
    int col = r / 64, k = r % 64;
    const float* W = (br == 0) ? W1 : (br == 1) ? W2 : W3;
    Wt[(size_t)br * 8192 + (size_t)col * 64 + k] = f2bf(W[k * 128 + col]);
}

// ---------------- WMMA GEMM: h[br] = X @ W_br  ([N,64]x[64,128]) ----------------
__global__ __launch_bounds__(256) void k_gemm(const __bf16* __restrict__ Xb,
                                              const __bf16* __restrict__ Wt,
                                              float* __restrict__ hfeat, int N) {
    int br   = blockIdx.y;
    int tile = blockIdx.x;               // 16-row tile
    int lane = threadIdx.x & 31;
    int w    = threadIdx.x >> 5;         // wave id = column tile 0..7
    int half = lane >> 4;                // 0/1
    int m    = lane & 15;

    int row = tile * 16 + m;
    if (row >= N) row = N - 1;
    const __bf16* arow = Xb + (size_t)row * 64;
    const __bf16* bcol = Wt + (size_t)br * 8192 + (size_t)(w * 16 + m) * 64;

    v8f acc = {};
#pragma unroll
    for (int ks = 0; ks < 2; ++ks) {
        v16bf a, b;
        // A 16x32 bf16 layout: lanes 0-15 hold K=0..7 (v0..3) and 16..23 (v4..7);
        // lanes 16-31 hold K=8..15 and 24..31.
        int ab0 = ks * 32 + half * 8;
        int ab1 = ks * 32 + 16 + half * 8;
#pragma unroll
        for (int i = 0; i < 8; ++i) { a[i] = arow[ab0 + i]; a[8 + i] = arow[ab1 + i]; }
        // B 32x16 bf16 layout: same K order as A (lanes 0-15 K=0..7,16..23; lanes 16-31 K=8..15,24..31); col = lane%16.
#pragma unroll
        for (int i = 0; i < 8; ++i) { b[i] = bcol[ab0 + i]; b[8 + i] = bcol[ab1 + i]; }
        acc = __builtin_amdgcn_wmma_f32_16x16x32_bf16(false, a, false, b,
                                                      (short)0, acc, false, false);
    }
    // C/D layout: VGPR r -> M = r + 8*half, col = lane%16
    int col = w * 16 + m;
    float* cbase = hfeat + (size_t)br * N * 128
                 + (size_t)(tile * 16 + half * 8) * 128 + col;
    if ((tile + 1) * 16 <= N) {
        // full tile: 8 unguarded stores, constant 512B immediate offsets
#pragma unroll
        for (int r = 0; r < 8; ++r) cbase[(size_t)r * 128] = acc[r];
    } else {
#pragma unroll
        for (int r = 0; r < 8; ++r) {
            int rm = tile * 16 + r + half * 8;
            if (rm < N) cbase[(size_t)r * 128] = acc[r];
        }
    }
}

// ---------------- per-node attention scores (layer 1) ----------------
__global__ void k_scores(const float* __restrict__ hfeat,
                         const float* as1, const float* ad1,
                         const float* as2, const float* ad2,
                         const float* as3, const float* ad3,
                         float* __restrict__ ssrc, float* __restrict__ sdst, int N) {
    int t = blockIdx.x * blockDim.x + threadIdx.x;
    if (t >= 3 * N * 8) return;
    int br = t / (N * 8);
    int rem = t % (N * 8);
    int n = rem / 8, hd = rem % 8;
    const float* as = (br == 0) ? as1 : (br == 1) ? as2 : as3;
    const float* ad = (br == 0) ? ad1 : (br == 1) ? ad2 : ad3;
    const float* hp = hfeat + (size_t)br * N * 128 + (size_t)n * 128 + hd * 16;
    float s0 = 0.f, s1 = 0.f;
#pragma unroll
    for (int c = 0; c < 16; ++c) { float v = hp[c]; s0 += v * as[hd * 16 + c]; s1 += v * ad[hd * 16 + c]; }
    ssrc[t] = s0; sdst[t] = s1;
}

// ---------------- init scratch ----------------
__global__ void k_init(float* emax, float* denom, float* xc,
                       float* emax2, float* den2, float* agg2, int N) {
    int t = blockIdx.x * blockDim.x + threadIdx.x;
    if (t < 3 * N * 8) { emax[t] = -1e30f; denom[t] = 0.f; }
    if (t < N * 384)   { xc[t] = 0.f; }
    if (t < 3 * N)     { emax2[t] = -1e30f; den2[t] = 0.f; agg2[t] = 0.f; }
}

// ---------------- layer-1 edge passes (gridDim.y = branch) ----------------
__global__ void k_edge_max(const int* A1, const int* A2, const int* A3,
                           const float* __restrict__ ssrc, const float* __restrict__ sdst,
                           float* __restrict__ emax, int N, int E) {
    int br = blockIdx.y;
    int e = blockIdx.x * blockDim.x + threadIdx.x;
    if (e >= E + N) return;
    const int* A = (br == 0) ? A1 : (br == 1) ? A2 : A3;
    int src, dst;
    if (e < E) { src = A[e]; dst = A[E + e]; } else { src = dst = e - E; }
    const float* ss = ssrc + (size_t)br * N * 8 + (size_t)src * 8;
    const float* sd = sdst + (size_t)br * N * 8 + (size_t)dst * 8;
    float* em = emax + (size_t)br * N * 8 + (size_t)dst * 8;
#pragma unroll
    for (int h = 0; h < 8; ++h) atomicMaxFloat(em + h, leaky(ss[h] + sd[h]));
}

__global__ void k_edge_sum(const int* A1, const int* A2, const int* A3,
                           const float* __restrict__ ssrc, const float* __restrict__ sdst,
                           const float* __restrict__ emax, float* __restrict__ denom,
                           int N, int E) {
    int br = blockIdx.y;
    int e = blockIdx.x * blockDim.x + threadIdx.x;
    if (e >= E + N) return;
    const int* A = (br == 0) ? A1 : (br == 1) ? A2 : A3;
    int src, dst;
    if (e < E) { src = A[e]; dst = A[E + e]; } else { src = dst = e - E; }
    const float* ss = ssrc + (size_t)br * N * 8 + (size_t)src * 8;
    const float* sd = sdst + (size_t)br * N * 8 + (size_t)dst * 8;
    const float* em = emax + (size_t)br * N * 8 + (size_t)dst * 8;
    float* dn = denom + (size_t)br * N * 8 + (size_t)dst * 8;
#pragma unroll
    for (int h = 0; h < 8; ++h)
        atomicAdd(dn + h, __expf(leaky(ss[h] + sd[h]) - em[h]));
}

// one thread per (edge, channel): coalesced 512B gather of h[src], atomic scatter to xc[dst]
__global__ void k_edge_agg(const int* A1, const int* A2, const int* A3,
                           const float* __restrict__ ssrc, const float* __restrict__ sdst,
                           const float* __restrict__ emax, const float* __restrict__ denom,
                           const float* __restrict__ hfeat, float* __restrict__ xc,
                           int N, int E) {
    int br = blockIdx.y;
    int t = blockIdx.x * blockDim.x + threadIdx.x;
    int Etot = E + N;
    if (t >= Etot * 128) return;
    int e  = t >> 7;
    int ch = t & 127;
    int hd = ch >> 4;
    const int* A = (br == 0) ? A1 : (br == 1) ? A2 : A3;
    int src, dst;
    if (e < E) { src = A[e]; dst = A[E + e]; } else { src = dst = e - E; }
    size_t b8 = (size_t)br * N * 8;
    float ev  = leaky(ssrc[b8 + (size_t)src * 8 + hd] + sdst[b8 + (size_t)dst * 8 + hd]);
    float ex  = __expf(ev - emax[b8 + (size_t)dst * 8 + hd]);
    float coef = ex / (denom[b8 + (size_t)dst * 8 + hd] + 1e-16f);
    float hv = hfeat[(size_t)br * N * 128 + (size_t)src * 128 + ch];
    atomicAdd(&xc[(size_t)dst * 384 + br * 128 + ch], coef * hv);
}

// ---------------- bias + relu into Xc ----------------
__global__ void k_bias_relu(float* __restrict__ xc, const float* b1, const float* b2,
                            const float* b3, int N) {
    int t = blockIdx.x * blockDim.x + threadIdx.x;
    if (t >= N * 384) return;
    int col = t % 384;
    int br = col / 128, cc = col % 128;
    const float* b = (br == 0) ? b1 : (br == 1) ? b2 : b3;
    xc[t] = fmaxf(xc[t] + b[cc], 0.f);
}

// ---------------- layer-2 projection: one wave per (node, branch) ----------------
__global__ void k_h2(const float* __restrict__ xc,
                     const float* W1, const float* W2, const float* W3,
                     const float* as1, const float* ad1,
                     const float* as2, const float* ad2,
                     const float* as3, const float* ad3,
                     float* __restrict__ h2, float* __restrict__ s2s,
                     float* __restrict__ s2d, int N) {
    int gw = (blockIdx.x * blockDim.x + threadIdx.x) >> 5;
    int lane = threadIdx.x & 31;
    if (gw >= 3 * N) return;
    int br = gw / N, n = gw % N;
    const float* W = (br == 0) ? W1 : (br == 1) ? W2 : W3;
    const float* row = xc + (size_t)n * 384;
    float acc = 0.f;
#pragma unroll
    for (int i = 0; i < 12; ++i) {
        int k = lane + i * 32;
        acc += row[k] * W[k];
    }
#pragma unroll
    for (int off = 16; off > 0; off >>= 1) acc += __shfl_xor(acc, off, 32);
    if (lane == 0) {
        const float* as = (br == 0) ? as1 : (br == 1) ? as2 : as3;
        const float* ad = (br == 0) ? ad1 : (br == 1) ? ad2 : ad3;
        h2[(size_t)br * N + n]  = acc;
        s2s[(size_t)br * N + n] = acc * as[0];
        s2d[(size_t)br * N + n] = acc * ad[0];
    }
}

// ---------------- layer-2 edge passes (scalar, H=1) ----------------
__global__ void k_edge2_max(const int* A1, const int* A2, const int* A3,
                            const float* __restrict__ s2s, const float* __restrict__ s2d,
                            float* __restrict__ emax2, int N, int E) {
    int br = blockIdx.y;
    int e = blockIdx.x * blockDim.x + threadIdx.x;
    if (e >= E + N) return;
    const int* A = (br == 0) ? A1 : (br == 1) ? A2 : A3;
    int src, dst;
    if (e < E) { src = A[e]; dst = A[E + e]; } else { src = dst = e - E; }
    atomicMaxFloat(&emax2[(size_t)br * N + dst],
                   leaky(s2s[(size_t)br * N + src] + s2d[(size_t)br * N + dst]));
}

__global__ void k_edge2_sum(const int* A1, const int* A2, const int* A3,
                            const float* __restrict__ s2s, const float* __restrict__ s2d,
                            const float* __restrict__ emax2, float* __restrict__ den2,
                            int N, int E) {
    int br = blockIdx.y;
    int e = blockIdx.x * blockDim.x + threadIdx.x;
    if (e >= E + N) return;
    const int* A = (br == 0) ? A1 : (br == 1) ? A2 : A3;
    int src, dst;
    if (e < E) { src = A[e]; dst = A[E + e]; } else { src = dst = e - E; }
    float ev = leaky(s2s[(size_t)br * N + src] + s2d[(size_t)br * N + dst]);
    atomicAdd(&den2[(size_t)br * N + dst], __expf(ev - emax2[(size_t)br * N + dst]));
}

__global__ void k_edge2_agg(const int* A1, const int* A2, const int* A3,
                            const float* __restrict__ s2s, const float* __restrict__ s2d,
                            const float* __restrict__ emax2, const float* __restrict__ den2,
                            const float* __restrict__ h2, float* __restrict__ agg2,
                            int N, int E) {
    int br = blockIdx.y;
    int e = blockIdx.x * blockDim.x + threadIdx.x;
    if (e >= E + N) return;
    const int* A = (br == 0) ? A1 : (br == 1) ? A2 : A3;
    int src, dst;
    if (e < E) { src = A[e]; dst = A[E + e]; } else { src = dst = e - E; }
    size_t bo = (size_t)br * N;
    float ev = leaky(s2s[bo + src] + s2d[bo + dst]);
    float ex = __expf(ev - emax2[bo + dst]);
    float coef = ex / (den2[bo + dst] + 1e-16f);
    atomicAdd(&agg2[bo + dst], coef * h2[bo + src]);
}

// ---------------- final linear ----------------
__global__ void k_final(const float* __restrict__ agg2,
                        const float* b1, const float* b2, const float* b3,
                        const float* lnw, const float* lnb,
                        float* __restrict__ out, int N) {
    int n = blockIdx.x * blockDim.x + threadIdx.x;
    if (n >= N) return;
    float y = (agg2[n] + b1[0]) * lnw[0]
            + (agg2[(size_t)N + n] + b2[0]) * lnw[1]
            + (agg2[(size_t)2 * N + n] + b3[0]) * lnw[2] + lnb[0];
    out[n] = y;
}

extern "C" void kernel_launch(void* const* d_in, const int* in_sizes, int n_in,
                              void* d_out, int out_size, void* d_ws, size_t ws_size,
                              hipStream_t stream) {
    const float* X  = (const float*)d_in[0];
    const int* A1   = (const int*)d_in[1];
    const int* A2   = (const int*)d_in[2];
    const int* A3   = (const int*)d_in[3];
    // d_in[4] = edge_feature (unused by reference)
    const float* W11 = (const float*)d_in[5];
    const float* as11 = (const float*)d_in[6];
    const float* ad11 = (const float*)d_in[7];
    const float* b11  = (const float*)d_in[8];
    const float* W12  = (const float*)d_in[9];
    const float* as12 = (const float*)d_in[10];
    const float* ad12 = (const float*)d_in[11];
    const float* b12  = (const float*)d_in[12];
    const float* W21  = (const float*)d_in[13];
    const float* as21 = (const float*)d_in[14];
    const float* ad21 = (const float*)d_in[15];
    const float* b21  = (const float*)d_in[16];
    const float* W22  = (const float*)d_in[17];
    const float* as22 = (const float*)d_in[18];
    const float* ad22 = (const float*)d_in[19];
    const float* b22  = (const float*)d_in[20];
    const float* W31  = (const float*)d_in[21];
    const float* as31 = (const float*)d_in[22];
    const float* ad31 = (const float*)d_in[23];
    const float* b31  = (const float*)d_in[24];
    const float* W32  = (const float*)d_in[25];
    const float* as32 = (const float*)d_in[26];
    const float* ad32 = (const float*)d_in[27];
    const float* b32  = (const float*)d_in[28];
    const float* lnw  = (const float*)d_in[29];
    const float* lnb  = (const float*)d_in[30];

    const int N = in_sizes[0] / 64;
    const int E = in_sizes[1] / 2;
    const int Etot = E + N;

    // ---- workspace carving (256B aligned chunks) ----
    size_t off = 0;
    char* base = (char*)d_ws;
    auto carve = [&](size_t bytes) -> char* {
        char* p = base + off;
        off += (bytes + 255) & ~(size_t)255;
        return p;
    };
    __bf16* Xb   = (__bf16*)carve((size_t)N * 64 * 2);
    __bf16* Wt   = (__bf16*)carve((size_t)3 * 128 * 64 * 2);
    float* hfeat = (float*)carve((size_t)3 * N * 128 * 4);
    float* ssrc  = (float*)carve((size_t)3 * N * 8 * 4);
    float* sdst  = (float*)carve((size_t)3 * N * 8 * 4);
    float* emax  = (float*)carve((size_t)3 * N * 8 * 4);
    float* denom = (float*)carve((size_t)3 * N * 8 * 4);
    float* xc    = (float*)carve((size_t)N * 384 * 4);
    float* h2    = (float*)carve((size_t)3 * N * 4);
    float* s2s   = (float*)carve((size_t)3 * N * 4);
    float* s2d   = (float*)carve((size_t)3 * N * 4);
    float* emax2 = (float*)carve((size_t)3 * N * 4);
    float* den2  = (float*)carve((size_t)3 * N * 4);
    float* agg2  = (float*)carve((size_t)3 * N * 4);
    (void)ws_size; (void)n_in;

    const int B = 256;
    // conversions
    k_cvt_x<<<(N * 64 + B - 1) / B, B, 0, stream>>>(X, Xb, N * 64);
    k_cvt_w<<<(3 * 128 * 64 + B - 1) / B, B, 0, stream>>>(W11, W21, W31, Wt);
    // WMMA GEMM: h[br] = X @ W_br
    dim3 gg((N + 15) / 16, 3);
    k_gemm<<<gg, B, 0, stream>>>(Xb, Wt, hfeat, N);
    // scores + init
    k_scores<<<(3 * N * 8 + B - 1) / B, B, 0, stream>>>(hfeat, as11, ad11, as21, ad21,
                                                        as31, ad31, ssrc, sdst, N);
    k_init<<<(N * 384 + B - 1) / B, B, 0, stream>>>(emax, denom, xc, emax2, den2, agg2, N);
    // layer-1 edge softmax + aggregation (3 branches via grid.y)
    dim3 ge((Etot + B - 1) / B, 3);
    k_edge_max<<<ge, B, 0, stream>>>(A1, A2, A3, ssrc, sdst, emax, N, E);
    k_edge_sum<<<ge, B, 0, stream>>>(A1, A2, A3, ssrc, sdst, emax, denom, N, E);
    dim3 ga(((size_t)Etot * 128 + B - 1) / B, 3);
    k_edge_agg<<<ga, B, 0, stream>>>(A1, A2, A3, ssrc, sdst, emax, denom, hfeat, xc, N, E);
    // bias + relu -> Xc
    k_bias_relu<<<(N * 384 + B - 1) / B, B, 0, stream>>>(xc, b11, b21, b31, N);
    // layer-2 projection (wave per node/branch)
    k_h2<<<((size_t)3 * N * 32 + B - 1) / B, B, 0, stream>>>(xc, W12, W22, W32,
                                                             as12, ad12, as22, ad22, as32, ad32,
                                                             h2, s2s, s2d, N);
    // layer-2 edge softmax + aggregation
    k_edge2_max<<<ge, B, 0, stream>>>(A1, A2, A3, s2s, s2d, emax2, N, E);
    k_edge2_sum<<<ge, B, 0, stream>>>(A1, A2, A3, s2s, s2d, emax2, den2, N, E);
    k_edge2_agg<<<ge, B, 0, stream>>>(A1, A2, A3, s2s, s2d, emax2, den2, h2, agg2, N, E);
    // final
    k_final<<<(N + B - 1) / B, B, 0, stream>>>(agg2, b12, b22, b32, lnw, lnb,
                                               (float*)d_out, N);
}
